// MHIM_9869834847083
// MI455X (gfx1250) — matrix operand order repaired and checked by reading, not compiled, not run on hardware
//
#include <hip/hip_runtime.h>

// ---------------- model constants ----------------
#define BATCH 2
#define CDIM  512
#define NSEQ  8192
#define NTOK  8193      // with cls token
#define NP    8448      // padded seq (multiple of 256)
#define PADR  255       // zero rows padded at FRONT
#define ML    256       // landmarks
#define LSEG  33        // rows per landmark (NP/ML)
#define NH    8
#define DH    64
#define BHD   16        // BATCH*NH
#define RK    33        // residual conv kernel
#define GRD   91        // ppeg grid
#define G2    8281      // 91*91

typedef __bf16 v16bf __attribute__((ext_vector_type(16)));
typedef float  v8f   __attribute__((ext_vector_type(8)));

// bf16 <-> f32 via explicit RNE bit ops (no native-cvt dependence)
__device__ inline __bf16 tobf(float x) {
  unsigned u = __float_as_uint(x);
  unsigned r = (u + 0x7fffu + ((u >> 16) & 1u)) >> 16;
  unsigned short s = (unsigned short)r;
  __bf16 o; __builtin_memcpy(&o, &s, 2); return o;
}
__device__ inline float frombf(__bf16 b) {
  unsigned short s; __builtin_memcpy(&s, &b, 2);
  return __uint_as_float(((unsigned)s) << 16);
}

// block-wide reductions (blockDim.x == 256)
__device__ inline float blk_sum(float v, float* red) {
  int t = threadIdx.x; red[t] = v; __syncthreads();
  for (int s = 128; s > 0; s >>= 1) { if (t < s) red[t] += red[t + s]; __syncthreads(); }
  float r = red[0]; __syncthreads(); return r;
}
__device__ inline float blk_max(float v, float* red) {
  int t = threadIdx.x; red[t] = v; __syncthreads();
  for (int s = 128; s > 0; s >>= 1) { if (t < s) red[t] = fmaxf(red[t], red[t + s]); __syncthreads(); }
  float r = red[0]; __syncthreads(); return r;
}

// ---------------- batched WMMA GEMM, 32x64 register-blocked ----------------
// C[M,N] = alpha * A[M,K] x B      (bf16 operands, f32 accumulate)
// BT=1: Bm is [N,K] row-major (C = A * Bm^T)  — "NT"
// BT=0: Bm is [K,N] row-major                 — "NN"
// One 32(M)x64(N) strip per wave: 2 A-frags x 4 B-frags -> 8 WMMAs / 32-K step.
// Requires M%32==0, N%64==0, K%32==0 (true for every call site here).
template<int BT>
__global__ void k_gemm(const __bf16* __restrict__ A, const __bf16* __restrict__ Bm,
                       float* Cf, __bf16* Cb,
                       int Md, int Nd, int Kd, int lda, int ldb, int ldc,
                       long long sA, long long sB, long long sC, float alpha)
{
  const int bh = blockIdx.z;
  const __bf16* Ab = A  + (size_t)bh * sA;
  const __bf16* Bb = Bm + (size_t)bh * sB;
  const int wave = threadIdx.x >> 5;
  const int lane = threadIdx.x & 31;
  const int nt = Nd >> 6;                                   // 64-wide N tiles
  const long long tile = (long long)blockIdx.x * (blockDim.x >> 5) + wave;
  if (tile >= (long long)(Md >> 5) * nt) return;            // wave-uniform: EXEC stays full
  const int tm = (int)(tile / nt), tn = (int)(tile % nt);
  const int r = lane & 15, hl = lane >> 4;

  const __bf16* arow0 = Ab + (size_t)(tm * 32 + r) * lda;
  const __bf16* arow1 = arow0 + (size_t)16 * lda;
  v8f acc[2][4] = {};
  for (int k0 = 0; k0 < Kd; k0 += 32) {
    v16bf a0, a1;
    // A 16x32 layout: lanes 0-15 -> K 0..15 ; lanes 16-31 -> K 16..31 (matches B)
#pragma unroll
    for (int e = 0; e < 8; ++e) {
      a0[e]     = arow0[k0 + hl * 16 + e];
      a0[e + 8] = arow0[k0 + hl * 16 + 8 + e];
      a1[e]     = arow1[k0 + hl * 16 + e];
      a1[e + 8] = arow1[k0 + hl * 16 + 8 + e];
    }
#pragma unroll
    for (int j = 0; j < 4; ++j) {
      v16bf b;
      if (BT) {  // B[k][n] = Bm[n][k]; lane n=r; element e -> k = k0 + hl*16 + e
        const __bf16* brow = Bb + (size_t)(tn * 64 + j * 16 + r) * ldb + k0 + hl * 16;
#pragma unroll
        for (int e = 0; e < 16; ++e) b[e] = brow[e];
      } else {   // B[k][n] = Bm[k][n]; strided in k, coalesced across lanes in n
        const __bf16* bcol = Bb + (size_t)(k0 + hl * 16) * ldb + tn * 64 + j * 16 + r;
#pragma unroll
        for (int e = 0; e < 16; ++e) b[e] = bcol[(size_t)e * ldb];
      }
      acc[0][j] = __builtin_amdgcn_wmma_f32_16x16x32_bf16(false, a0, false, b, (short)0,
                                                          acc[0][j], false, false);
      acc[1][j] = __builtin_amdgcn_wmma_f32_16x16x32_bf16(false, a1, false, b, (short)0,
                                                          acc[1][j], false, false);
    }
  }
  // C/D 16x16 f32 layout: VGPR e -> M = e + 8*half ; N = lane&15
#pragma unroll
  for (int i = 0; i < 2; ++i) {
#pragma unroll
    for (int j = 0; j < 4; ++j) {
      const int n = tn * 64 + j * 16 + r;
#pragma unroll
      for (int e = 0; e < 8; ++e) {
        const int m = tm * 32 + i * 16 + hl * 8 + e;
        const float v = acc[i][j][e] * alpha;
        const size_t idx = (size_t)bh * sC + (size_t)m * ldc + n;
        if (Cf) Cf[idx] = v;
        if (Cb) Cb[idx] = tobf(v);
      }
    }
  }
}

// ---------------- elementwise / reduction kernels ----------------
__global__ void k_f2b(const float* in, __bf16* out, long long n) {
  long long i = (long long)blockIdx.x * 256 + threadIdx.x;
  if (i < n) out[i] = tobf(in[i]);
}
__global__ void k_b2f(const __bf16* in, float* out, long long n) {
  long long i = (long long)blockIdx.x * 256 + threadIdx.x;
  if (i < n) out[i] = frombf(in[i]);
}

// h[b, 0, :] = cls ; h[b, 1+i, :] = x[b, i, :]
__global__ void k_init_h(const float* x, const float* cls, float* h) {
  long long idx = (long long)blockIdx.x * 256 + threadIdx.x;
  if (idx >= (long long)BATCH * NTOK * CDIM) return;
  int c = (int)(idx % CDIM); long long t = idx / CDIM;
  int i = (int)(t % NTOK);   int b = (int)(t / NTOK);
  h[idx] = (i == 0) ? cls[c] : x[((long long)b * NSEQ + i - 1) * CDIM + c];
}

// LayerNorm rows of h, write bf16 into padded [B, NP, C] (front PADR rows zero)
__global__ void k_ln_pad(const float* h, const float* w, const float* bb, __bf16* out) {
  __shared__ float red[256];
  int bid = blockIdx.x; int i = bid % NP; int b = bid / NP;
  int tid = threadIdx.x;
  __bf16* orow = out + ((size_t)b * NP + i) * CDIM;
  if (i < PADR) { orow[tid] = tobf(0.f); orow[tid + 256] = tobf(0.f); return; }
  const float* x = h + ((size_t)b * NTOK + (i - PADR)) * CDIM;
  float x0 = x[tid], x1 = x[tid + 256];
  float mu  = blk_sum(x0 + x1, red) * (1.f / CDIM);
  float d0 = x0 - mu, d1 = x1 - mu;
  float var = blk_sum(d0 * d0 + d1 * d1, red) * (1.f / CDIM);
  float rs = rsqrtf(var + 1e-5f);
  orow[tid]       = tobf(d0 * rs * w[tid]       + bb[tid]);
  orow[tid + 256] = tobf(d1 * rs * w[tid + 256] + bb[tid + 256]);
}

// split qkv f32 [B,NP,1536] into per-head bf16 [BH,NP,64]; q scaled by d^-0.5 = 1/8
__global__ void k_pack_qkv(const float* qkv, __bf16* q, __bf16* k, __bf16* v) {
  long long idx = (long long)blockIdx.x * 256 + threadIdx.x;
  if (idx >= (long long)BATCH * NP * CDIM) return;
  int c = (int)(idx % CDIM); long long t = idx / CDIM;
  int i = (int)(t % NP);     int b = (int)(t / NP);
  int h = c >> 6, d = c & 63;
  const float* base = qkv + ((size_t)b * NP + i) * (3 * CDIM) + c;
  size_t o = (((size_t)(b * NH + h)) * NP + i) * DH + d;
  q[o] = tobf(base[0]        * 0.125f);
  k[o] = tobf(base[CDIM]);
  v[o] = tobf(base[2 * CDIM]);
}

// landmark means: block per (bh, landmark), 64 threads (one per d)
__global__ void k_landmarks(const float* qkv, __bf16* ql, __bf16* kl) {
  int bid = blockIdx.x; int i = bid % ML; int bh = bid / ML;
  int h = bh % NH, b = bh / NH, d = threadIdx.x;
  const float* base = qkv + ((size_t)b * NP + (size_t)i * LSEG) * (3 * CDIM) + h * DH + d;
  float sq = 0.f, sk = 0.f;
  for (int j = 0; j < LSEG; ++j) { sq += base[(size_t)j * 3 * CDIM]; sk += base[(size_t)j * 3 * CDIM + CDIM]; }
  size_t o = ((size_t)bh * ML + i) * DH + d;
  ql[o] = tobf(sq * (0.125f / (float)LSEG));
  kl[o] = tobf(sk * (1.0f   / (float)LSEG));
}

// in-place row softmax on bf16 (fp32 statistics)
__global__ void k_softmax(__bf16* buf, int len) {
  __shared__ float red[256];
  __bf16* row = buf + (size_t)blockIdx.x * len;
  int tid = threadIdx.x;
  float mx = -3.0e38f;
  for (int j = tid; j < len; j += 256) mx = fmaxf(mx, frombf(row[j]));
  mx = blk_max(mx, red);
  float sm = 0.f;
  for (int j = tid; j < len; j += 256) sm += __expf(frombf(row[j]) - mx);
  sm = blk_sum(sm, red);
  float inv = 1.f / sm;
  for (int j = tid; j < len; j += 256) row[j] = tobf(__expf(frombf(row[j]) - mx) * inv);
}

// Moore-Penrose init: z = x^T / (max_i sum_j|x| * max_j sum_i|x|) ; block per bh matrix
__global__ void k_pinv_init(const float* x2, __bf16* z) {
  __shared__ float red[256];
  const float* x = x2 + (size_t)blockIdx.x * ML * ML;
  __bf16* zo = z + (size_t)blockIdx.x * ML * ML;
  int tid = threadIdx.x;
  float rs = 0.f, cs = 0.f;
  for (int j = 0; j < ML; ++j) { rs += fabsf(x[(size_t)tid * ML + j]); cs += fabsf(x[(size_t)j * ML + tid]); }
  float colm = blk_max(rs, red);
  float rowm = blk_max(cs, red);
  float inv = 1.f / (colm * rowm);
  for (int i = 0; i < ML; ++i) zo[(size_t)tid * ML + i] = tobf(x[(size_t)i * ML + tid] * inv);
}

// out = c*I - in  (in: f32 ptr or bf16 ptr, one null), batched 256x256
__global__ void k_ident_sub(const float* inf, const __bf16* inb, __bf16* out, float c, long long n) {
  long long idx = (long long)blockIdx.x * 256 + threadIdx.x;
  if (idx >= n) return;
  int mi = (int)(idx % (ML * ML));
  float v = inf ? inf[idx] : frombf(inb[idx]);
  out[idx] = tobf((((mi >> 8) == (mi & 255)) ? c : 0.f) - v);
}

// depthwise seq-conv residual (kernel 33x1 per head), add into outh [BH,NP,64]
__global__ void k_res_add(const float* qkv, const float* rw, float* outh) {
  long long idx = (long long)blockIdx.x * 256 + threadIdx.x;
  if (idx >= (long long)BHD * NP * DH) return;
  int d = (int)(idx & 63); long long t = idx >> 6;
  int i = (int)(t % NP); int bh = (int)(t / NP);
  int h = bh % NH, b = bh / NH;
  float acc = outh[idx];
  const float* vbase = qkv + 2 * CDIM + h * DH + d;
  for (int tt = 0; tt < RK; ++tt) {
    int s = i + tt - RK / 2;
    if (s >= 0 && s < NP) acc += rw[h * RK + tt] * vbase[((size_t)b * NP + s) * (3 * CDIM)];
  }
  outh[idx] = acc;
}

// [BH,NP,64] f32 -> [B,NP,512] bf16 (head concat)
__global__ void k_pack_att(const float* outh, __bf16* att) {
  long long idx = (long long)blockIdx.x * 256 + threadIdx.x;
  if (idx >= (long long)BATCH * NP * CDIM) return;
  int c = (int)(idx % CDIM); long long t = idx / CDIM;
  int i = (int)(t % NP); int b = (int)(t / NP);
  att[idx] = tobf(outh[(((size_t)(b * NH + (c >> 6))) * NP + i) * DH + (c & 63)]);
}

// h += y[:, PADR:, :] + out_b  (residual + bias; drops pad rows)
__global__ void k_epilogue(float* h, const float* y, const float* ob) {
  long long idx = (long long)blockIdx.x * 256 + threadIdx.x;
  if (idx >= (long long)BATCH * NTOK * CDIM) return;
  int c = (int)(idx % CDIM); long long t = idx / CDIM;
  int i = (int)(t % NTOK); int b = (int)(t / NTOK);
  h[idx] += y[((size_t)b * NP + PADR + i) * CDIM + c] + ob[c];
}

// ---------------- PPEG ----------------
__global__ void k_to_grid(const float* h, float* feat) {
  long long idx = (long long)blockIdx.x * 256 + threadIdx.x;
  if (idx >= (long long)BATCH * G2 * CDIM) return;
  int c = (int)(idx % CDIM); long long t = idx / CDIM;
  int p = (int)(t % G2); int b = (int)(t / G2);
  int seq = (p < NSEQ) ? p : p - NSEQ;           // wrap-concat of first `add` rows
  feat[((size_t)b * CDIM + c) * G2 + p] = h[((size_t)b * NTOK + 1 + seq) * CDIM + c];
}

__global__ void k_ppeg(const float* feat, const float* w7, const float* b7,
                       const float* w5, const float* b5, const float* w3, const float* b3,
                       float* h) {
  long long idx = (long long)blockIdx.x * 256 + threadIdx.x;
  if (idx >= (long long)BATCH * CDIM * G2) return;
  int p = (int)(idx % G2); long long t = idx / G2;
  int c = (int)(t % CDIM); int b = (int)(t / CDIM);
  if (p >= NSEQ) return;                          // only first 8192 written back
  int gy = p / GRD, gx = p % GRD;
  const float* f = feat + ((size_t)b * CDIM + c) * G2;
  float acc = f[p] + b7[c] + b5[c] + b3[c];
  for (int ky = 0; ky < 7; ++ky) { int iy = gy + ky - 3; if (iy < 0 || iy >= GRD) continue;
    for (int kx = 0; kx < 7; ++kx) { int ix = gx + kx - 3; if (ix < 0 || ix >= GRD) continue;
      acc += w7[c * 49 + ky * 7 + kx] * f[iy * GRD + ix]; } }
  for (int ky = 0; ky < 5; ++ky) { int iy = gy + ky - 2; if (iy < 0 || iy >= GRD) continue;
    for (int kx = 0; kx < 5; ++kx) { int ix = gx + kx - 2; if (ix < 0 || ix >= GRD) continue;
      acc += w5[c * 25 + ky * 5 + kx] * f[iy * GRD + ix]; } }
  for (int ky = 0; ky < 3; ++ky) { int iy = gy + ky - 1; if (iy < 0 || iy >= GRD) continue;
    for (int kx = 0; kx < 3; ++kx) { int ix = gx + kx - 1; if (ix < 0 || ix >= GRD) continue;
      acc += w3[c * 9 + ky * 3 + kx] * f[iy * GRD + ix]; } }
  h[((size_t)b * NTOK + 1 + p) * CDIM + c] = acc;
}

// final LN of cls row -> d_out [B, 512] f32
__global__ void k_final_ln(const float* h, const float* w, const float* bb, float* out) {
  __shared__ float red[256];
  int b = blockIdx.x, tid = threadIdx.x;
  const float* x = h + (size_t)b * NTOK * CDIM;   // row 0
  float x0 = x[tid], x1 = x[tid + 256];
  float mu  = blk_sum(x0 + x1, red) * (1.f / CDIM);
  float d0 = x0 - mu, d1 = x1 - mu;
  float var = blk_sum(d0 * d0 + d1 * d1, red) * (1.f / CDIM);
  float rs = rsqrtf(var + 1e-5f);
  out[(size_t)b * CDIM + tid]       = d0 * rs * w[tid]       + bb[tid];
  out[(size_t)b * CDIM + tid + 256] = d1 * rs * w[tid + 256] + bb[tid + 256];
}

// ---------------- host orchestration ----------------
extern "C" void kernel_launch(void* const* d_in, const int* in_sizes, int n_in,
                              void* d_out, int out_size, void* d_ws, size_t ws_size,
                              hipStream_t stream) {
  (void)in_sizes; (void)n_in; (void)out_size; (void)ws_size;
  const float* X    = (const float*)d_in[0];
  const float* CLS  = (const float*)d_in[1];
  const float* LW[2]   = { (const float*)d_in[2],  (const float*)d_in[8]  };
  const float* LB[2]   = { (const float*)d_in[3],  (const float*)d_in[9]  };
  const float* QKVW[2] = { (const float*)d_in[4],  (const float*)d_in[10] };
  const float* OUTW[2] = { (const float*)d_in[5],  (const float*)d_in[11] };
  const float* OUTB[2] = { (const float*)d_in[6],  (const float*)d_in[12] };
  const float* RESW[2] = { (const float*)d_in[7],  (const float*)d_in[13] };
  const float* W7 = (const float*)d_in[14]; const float* B7 = (const float*)d_in[15];
  const float* W5 = (const float*)d_in[16]; const float* B5 = (const float*)d_in[17];
  const float* W3 = (const float*)d_in[18]; const float* B3 = (const float*)d_in[19];
  const float* NW = (const float*)d_in[20]; const float* NB = (const float*)d_in[21];

  // ---- workspace carve (256B aligned bump allocator) ----
  size_t off = 0;
  auto carve = [&](size_t bytes) -> char* {
    char* r = (char*)d_ws + off; off = (off + bytes + 255) & ~(size_t)255; return r;
  };
  float*  h    = (float*) carve((size_t)BATCH * NTOK * CDIM * 4);
  __bf16* xln  = (__bf16*)carve((size_t)BATCH * NP * CDIM * 2);
  __bf16* wqkv = (__bf16*)carve((size_t)3 * CDIM * CDIM * 2);
  __bf16* wout = (__bf16*)carve((size_t)CDIM * CDIM * 2);
  float*  qkv  = (float*) carve((size_t)BATCH * NP * 3 * CDIM * 4);   // >= feat size
  float*  feat = qkv;                                                  // alias: ppeg runs when qkv is dead
  __bf16* qb   = (__bf16*)carve((size_t)BHD * NP * DH * 2);
  __bf16* kb   = (__bf16*)carve((size_t)BHD * NP * DH * 2);
  __bf16* vb   = (__bf16*)carve((size_t)BHD * NP * DH * 2);
  __bf16* qlb  = (__bf16*)carve((size_t)BHD * ML * DH * 2);
  __bf16* klb  = (__bf16*)carve((size_t)BHD * ML * DH * 2);
  __bf16* s1   = (__bf16*)carve((size_t)BHD * NP * ML * 2);           // attn1 (in-place softmax)
  float*  yf   = (float*)s1;                                           // alias: y computed after attn1 dead
  __bf16* s3   = (__bf16*)carve((size_t)BHD * ML * NP * 2);           // attn3
  __bf16* s2   = (__bf16*)carve((size_t)BHD * ML * ML * 2);           // attn2
  float*  x2   = (float*) carve((size_t)BHD * ML * ML * 4);
  __bf16* za   = (__bf16*)carve((size_t)BHD * ML * ML * 2);
  __bf16* zb2  = (__bf16*)carve((size_t)BHD * ML * ML * 2);
  __bf16* xzb  = (__bf16*)carve((size_t)BHD * ML * ML * 2);
  __bf16* wb   = (__bf16*)carve((size_t)BHD * ML * ML * 2);
  float*  tmpf = (float*) carve((size_t)BHD * ML * ML * 4);
  __bf16* t1   = (__bf16*)carve((size_t)BHD * ML * DH * 2);
  __bf16* t2b  = (__bf16*)carve((size_t)BHD * ML * DH * 2);
  float*  outh = (float*) carve((size_t)BHD * NP * DH * 4);
  __bf16* att  = (__bf16*)carve((size_t)BATCH * NP * CDIM * 2);

  auto nb256 = [](long long n) { return dim3((unsigned)((n + 255) / 256)); };
  auto gemm = [&](int BT, const __bf16* A, const __bf16* Bm, float* Cf, __bf16* Cb,
                  int Md, int Nd, int Kd, int lda, int ldb, int ldc,
                  long long sA, long long sB, long long sC, int batch, float alpha) {
    long long tiles = (long long)(Md / 32) * (Nd / 64);   // 32x64 strip per wave
    dim3 g((unsigned)((tiles + 7) / 8), 1, (unsigned)batch);
    if (BT) k_gemm<1><<<g, 256, 0, stream>>>(A, Bm, Cf, Cb, Md, Nd, Kd, lda, ldb, ldc, sA, sB, sC, alpha);
    else    k_gemm<0><<<g, 256, 0, stream>>>(A, Bm, Cf, Cb, Md, Nd, Kd, lda, ldb, ldc, sA, sB, sC, alpha);
  };

  auto layer = [&](int li) {
    const long long mm = (long long)ML * ML;
    k_f2b<<<nb256(3LL * CDIM * CDIM), 256, 0, stream>>>(QKVW[li], wqkv, 3LL * CDIM * CDIM);
    k_f2b<<<nb256((long long)CDIM * CDIM), 256, 0, stream>>>(OUTW[li], wout, (long long)CDIM * CDIM);
    k_ln_pad<<<BATCH * NP, 256, 0, stream>>>(h, LW[li], LB[li], xln);
    // qkv = xln @ qkv_w^T   (NT, weights shared across batch: sB=0)
    gemm(1, xln, wqkv, qkv, nullptr, NP, 3 * CDIM, CDIM, CDIM, CDIM, 3 * CDIM,
         (long long)NP * CDIM, 0, (long long)NP * 3 * CDIM, BATCH, 1.f);
    k_pack_qkv<<<nb256((long long)BATCH * NP * CDIM), 256, 0, stream>>>(qkv, qb, kb, vb);
    k_landmarks<<<BHD * ML, 64, 0, stream>>>(qkv, qlb, klb);
    // attention logits (NT, batched over 16 bh)
    gemm(1, qb,  klb, nullptr, s1, NP, ML, DH, DH, DH, ML,
         (long long)NP * DH, (long long)ML * DH, (long long)NP * ML, BHD, 1.f);
    gemm(1, qlb, klb, nullptr, s2, ML, ML, DH, DH, DH, ML,
         (long long)ML * DH, (long long)ML * DH, mm, BHD, 1.f);
    gemm(1, qlb, kb,  nullptr, s3, ML, NP, DH, DH, DH, NP,
         (long long)ML * DH, (long long)NP * DH, (long long)ML * NP, BHD, 1.f);
    k_softmax<<<BHD * NP, 256, 0, stream>>>(s1, ML);
    k_softmax<<<BHD * ML, 256, 0, stream>>>(s2, ML);
    k_softmax<<<BHD * ML, 256, 0, stream>>>(s3, NP);
    // Newton-Schulz pseudo-inverse of attn2 (bf16 GEMMs, f32 fix-ups)
    k_b2f<<<nb256(BHD * mm), 256, 0, stream>>>(s2, x2, BHD * mm);
    k_pinv_init<<<BHD, 256, 0, stream>>>(x2, za);
    __bf16 *zc = za, *zn = zb2;
    for (int it = 0; it < 6; ++it) {
      gemm(0, s2, zc, nullptr, xzb, ML, ML, ML, ML, ML, ML, mm, mm, mm, BHD, 1.f);   // XZ
      k_ident_sub<<<nb256(BHD * mm), 256, 0, stream>>>(nullptr, xzb, wb, 7.f, BHD * mm);
      gemm(0, xzb, wb, tmpf, nullptr, ML, ML, ML, ML, ML, ML, mm, mm, mm, BHD, 1.f); // XZ*W1
      k_ident_sub<<<nb256(BHD * mm), 256, 0, stream>>>(tmpf, nullptr, wb, 15.f, BHD * mm);
      gemm(0, xzb, wb, tmpf, nullptr, ML, ML, ML, ML, ML, ML, mm, mm, mm, BHD, 1.f); // XZ*W2
      k_ident_sub<<<nb256(BHD * mm), 256, 0, stream>>>(tmpf, nullptr, wb, 13.f, BHD * mm);
      gemm(0, zc, wb, nullptr, zn, ML, ML, ML, ML, ML, ML, mm, mm, mm, BHD, 0.25f);  // z' = 0.25*z*W3
      __bf16* t = zc; zc = zn; zn = t;
    }
    // out = attn1 @ (z @ (attn3 @ v)) + residual conv
    gemm(0, s3, vb, nullptr, t1, ML, DH, NP, NP, DH, DH,
         (long long)ML * NP, (long long)NP * DH, (long long)ML * DH, BHD, 1.f);
    gemm(0, zc, t1, nullptr, t2b, ML, DH, ML, ML, DH, DH,
         mm, (long long)ML * DH, (long long)ML * DH, BHD, 1.f);
    gemm(0, s1, t2b, outh, nullptr, NP, DH, ML, ML, DH, DH,
         (long long)NP * ML, (long long)ML * DH, (long long)NP * DH, BHD, 1.f);
    k_res_add<<<nb256((long long)BHD * NP * DH), 256, 0, stream>>>(qkv, RESW[li], outh);
    k_pack_att<<<nb256((long long)BATCH * NP * CDIM), 256, 0, stream>>>(outh, att);
    // y = att @ out_w^T ; then h += y[:, PADR:, :] + out_b
    gemm(1, att, wout, yf, nullptr, NP, CDIM, CDIM, CDIM, CDIM, CDIM,
         (long long)NP * CDIM, 0, (long long)NP * CDIM, BATCH, 1.f);
    k_epilogue<<<nb256((long long)BATCH * NTOK * CDIM), 256, 0, stream>>>(h, yf, OUTB[li]);
  };

  k_init_h<<<nb256((long long)BATCH * NTOK * CDIM), 256, 0, stream>>>(X, CLS, h);
  layer(0);
  k_to_grid<<<nb256((long long)BATCH * G2 * CDIM), 256, 0, stream>>>(h, feat);
  k_ppeg<<<nb256((long long)BATCH * CDIM * G2), 256, 0, stream>>>(feat, W7, B7, W5, B5, W3, B3, h);
  layer(1);
  k_final_ln<<<BATCH, 256, 0, stream>>>(h, NW, NB, (float*)d_out);
}
